// Gpt_25872882992060
// MI455X (gfx1250) — matrix operand order repaired and checked by reading, not compiled, not run on hardware
//
#include <hip/hip_runtime.h>
#include <hip/hip_bf16.h>

// ---------------- model constants ----------------
#define Vn   32000
#define Dm   1024
#define Hn   16
#define HDn  64
#define NLn  6
#define DFFn 4096
#define Sn   2048
#define Bn   2
#define Mtok (Bn * Sn)   // 4096 tokens

typedef unsigned short bf16x;          // raw bf16 bits
typedef unsigned int   u32;
typedef unsigned long long u64;

typedef __attribute__((ext_vector_type(16))) __bf16 v16bf;
typedef __attribute__((ext_vector_type(8)))  float  v8f;
typedef __attribute__((ext_vector_type(8)))  u32    v8u;
typedef __attribute__((ext_vector_type(4)))  u32    v4u;
typedef __attribute__((ext_vector_type(8)))  int    v8i;
typedef __attribute__((ext_vector_type(4)))  int    v4i;

// Tensor Data Mover path (gfx1250). Guarded so absence of the builtin (or the
// host pass) falls back to the staged register copy path.
#if defined(__AMDGCN__) && __has_builtin(__builtin_amdgcn_tensor_load_to_lds) && \
    __has_builtin(__builtin_amdgcn_s_wait_tensorcnt)
#define USE_TDM 1
#else
#define USE_TDM 0
#endif

static __device__ __forceinline__ bf16x f2bf(float f) {
    u32 u = __builtin_bit_cast(u32, f);
    u32 r = (u + 0x7FFFu + ((u >> 16) & 1u)) >> 16;   // round-to-nearest-even
    return (bf16x)r;
}
static __device__ __forceinline__ float bf2f(bf16x h) {
    u32 u = ((u32)h) << 16;
    return __builtin_bit_cast(float, u);
}
// ISA 16-bit A-fragment K offset for vgpr v, lane-half lh (16x16x32)
static __device__ __forceinline__ int a_koff(int v, int lh) {
    return (v < 4 ? 2 * v : 16 + 2 * (v - 4)) + 8 * lh;
}

// 2-D tile load global->LDS via TDM (D# per CDNA5 ISA ch.8).
// tile_w elems (2B each) per row, tile_h rows, global row pitch row_stride elems.
// pad_i/pad_a: LDS padding codes (interval 2<<code DWORDs, amount code+1 DWORDs).
static __device__ __forceinline__ void tdm_load_2d(u32 lds_byte_addr, const void* gptr,
                                                   u32 tile_w, u32 tile_h,
                                                   u32 tensor_w, u32 tensor_h,
                                                   u32 row_stride, u32 pad_i, u32 pad_a) {
#if USE_TDM
    const u64 ga = (u64)gptr;
    v4u g0;
    g0[0] = 1u;                                            // count=1 (valid user D#)
    g0[1] = lds_byte_addr;                                 // lds_addr
    g0[2] = (u32)(ga & 0xFFFFFFFFull);                     // global_addr[31:0]
    g0[3] = (u32)((ga >> 32) & 0x1FFFFFFull) | (2u << 30); // global_addr[56:32] | type=2
    v8i g1;
    g1[0] = (int)((1u << 16) | (1u << 20) | (pad_i << 22) | (pad_a << 25)); // 2B elems, pad
    g1[1] = (int)((tensor_w & 0xFFFFu) << 16);                              // dim0[15:0]
    g1[2] = (int)(((tensor_w >> 16) & 0xFFFFu) | ((tensor_h & 0xFFFFu) << 16));
    g1[3] = (int)(((tensor_h >> 16) & 0xFFFFu) | ((tile_w & 0xFFFFu) << 16)); // tile_dim0
    g1[4] = (int)(tile_h & 0xFFFFu);                                        // tile_dim1
    g1[5] = (int)row_stride;                                                // dim0_stride lo
    g1[6] = 0;
    g1[7] = 0;
    v4i g2; g2[0] = 0; g2[1] = 0; g2[2] = 0; g2[3] = 0;
    v4i g3; g3[0] = 0; g3[1] = 0; g3[2] = 0; g3[3] = 0;
#if __clang_major__ >= 23
    v8i g4; g4[0]=0; g4[1]=0; g4[2]=0; g4[3]=0; g4[4]=0; g4[5]=0; g4[6]=0; g4[7]=0;
    __builtin_amdgcn_tensor_load_to_lds(g0, g1, g2, g3, g4, 0);
#else
    __builtin_amdgcn_tensor_load_to_lds(g0, g1, g2, g3, 0);
#endif
#else
    (void)lds_byte_addr; (void)gptr; (void)tile_w; (void)tile_h;
    (void)tensor_w; (void)tensor_h; (void)row_stride; (void)pad_i; (void)pad_a;
#endif
}
static __device__ __forceinline__ void tdm_wait() {
#if USE_TDM
    __builtin_amdgcn_s_wait_tensorcnt(0);
#endif
}

// ---------------- fp32 -> bf16 weight conversion ----------------
__global__ __launch_bounds__(256) void k_cvt(const float* __restrict__ in,
                                             bf16x* __restrict__ out, int n) {
    int i = blockIdx.x * 256 + threadIdx.x;
    int stride = gridDim.x * 256;
    for (; i < n; i += stride) out[i] = f2bf(in[i]);
}

// ---------------- embedding + positional encoding ----------------
__global__ __launch_bounds__(256) void k_embed(const int* __restrict__ tok,
                                               const float* __restrict__ emb,
                                               const float* __restrict__ pos,
                                               float* __restrict__ X) {
    int i = blockIdx.x * 256 + threadIdx.x;
    int stride = gridDim.x * 256;
    const int total = Mtok * Dm;
    for (; i < total; i += stride) {
        int t = i >> 10;
        int d = i & 1023;
        int s = t & (Sn - 1);
        X[i] = emb[(size_t)tok[t] * Dm + d] * 32.0f + pos[(size_t)s * Dm + d];
    }
}

// ---------------- LayerNorm (torch-style: std with ddof=1, /(std+1e-5)) ----------------
__global__ __launch_bounds__(256) void k_ln(const float* __restrict__ X,
                                            const float* __restrict__ gk,
                                            const float* __restrict__ gb,
                                            bf16x* __restrict__ out) {
    __shared__ float sbuf[256];
    const int t = blockIdx.x;
    const int tid = threadIdx.x;
    const float* x = X + (size_t)t * Dm;

    float v[4];
    float s = 0.f;
#pragma unroll
    for (int i = 0; i < 4; ++i) { v[i] = x[tid + 256 * i]; s += v[i]; }
    sbuf[tid] = s; __syncthreads();
    for (int o = 128; o > 0; o >>= 1) { if (tid < o) sbuf[tid] += sbuf[tid + o]; __syncthreads(); }
    const float mean = sbuf[0] * (1.0f / (float)Dm);
    __syncthreads();

    float s2 = 0.f;
#pragma unroll
    for (int i = 0; i < 4; ++i) { float d = v[i] - mean; s2 += d * d; }
    sbuf[tid] = s2; __syncthreads();
    for (int o = 128; o > 0; o >>= 1) { if (tid < o) sbuf[tid] += sbuf[tid + o]; __syncthreads(); }
    const float stdv = sqrtf(sbuf[0] * (1.0f / (float)(Dm - 1)));   // ddof = 1
    const float inv = 1.0f / (stdv + 1e-5f);

#pragma unroll
    for (int i = 0; i < 4; ++i) {
        int d = tid + 256 * i;
        out[(size_t)t * Dm + d] = f2bf(gk[d] * ((v[i] - mean) * inv) + gb[d]);
    }
}

// ---------------- bf16 WMMA GEMM, double-buffered LDS, TDM A-tile ----------------
// C[M,N] = A[M,K] * W[N,K]^T (+bias)(+silu)(+resid)
// Block: 256 threads = 8 wave32 waves. Block tile 128x128, wave tile 32x64, K step 32.
#define GEMM_FLAG_SILU 1

__global__ __launch_bounds__(256) void k_gemm(const bf16x* __restrict__ A,
                                              const bf16x* __restrict__ W,
                                              const float* __restrict__ bias,
                                              const float* __restrict__ resid,
                                              float* __restrict__ outF,
                                              bf16x* __restrict__ outB,
                                              int M, int N, int K, int flags) {
    __shared__ bf16x At[2][128 * 34];   // stride 34 == TDM pad (64B rows + 4B pad)
    __shared__ bf16x Bt[2][32 * 130];   // W transposed [k][n], stride 130

    const int tid  = threadIdx.x;
    const int wid  = tid >> 5;
    const int lane = tid & 31;
    const int bm = blockIdx.y * 128;
    const int bn = blockIdx.x * 128;
    const int wm = (wid & 3) * 32;
    const int wn = (wid >> 2) * 64;
    const int m16 = lane & 15;
    const int lh  = lane >> 4;

    v8f acc[2][4];
#pragma unroll
    for (int i = 0; i < 2; ++i)
#pragma unroll
        for (int j = 0; j < 4; ++j)
#pragma unroll
            for (int r = 0; r < 8; ++r) acc[i][j][r] = 0.0f;

    uint4 dw[2];
#if !USE_TDM
    uint4 da[2];
#endif
    auto gload = [&](int k0) {
#pragma unroll
        for (int q = 0; q < 2; ++q) {
            const int c = tid + 256 * q;
            const int r = c >> 2, col = (c & 3) * 8;
#if !USE_TDM
            da[q] = *(const uint4*)(A + (size_t)(bm + r) * K + k0 + col);
#endif
            dw[q] = *(const uint4*)(W + (size_t)(bn + r) * K + k0 + col);
        }
    };
    auto sstore = [&](int buf) {
#pragma unroll
        for (int q = 0; q < 2; ++q) {
            const int c = tid + 256 * q;
            const int r = c >> 2, col = (c & 3) * 8;
#if !USE_TDM
            u32* dst = (u32*)&At[buf][r * 34 + col];
            dst[0] = da[q].x; dst[1] = da[q].y; dst[2] = da[q].z; dst[3] = da[q].w;
#endif
            const bf16x* wp = (const bf16x*)&dw[q];
#pragma unroll
            for (int i = 0; i < 8; ++i) Bt[buf][(col + i) * 130 + r] = wp[i];
        }
    };
    auto tdmA = [&](int buf, int k0) {
#if USE_TDM
        if (tid < 32)  // one TDM issue per block (wave 0)
            tdm_load_2d((u32)(size_t)&At[buf][0], A + (size_t)bm * K + k0,
                        32, 128, (u32)K, 128, (u32)K, /*pad 64B+4B*/ 3, 0);
#else
        (void)buf; (void)k0;
#endif
    };

    tdmA(0, 0);
    gload(0);
    sstore(0);
    tdm_wait();
    __syncthreads();
    int cur = 0;

    for (int k0 = 0; k0 < K; k0 += 32) {
        const bool more = (k0 + 32 < K);
        if (more) { tdmA(cur ^ 1, k0 + 32); gload(k0 + 32); }
        if (k0 + 64 < K) {
            const int pr = tid >> 1;
            __builtin_prefetch(A + (size_t)(bm + pr) * K + k0 + 64, 0, 1);
            __builtin_prefetch(W + (size_t)(bn + pr) * K + k0 + 64, 0, 1);
        }

        v8u afr[2], bfr[4];
#pragma unroll
        for (int i = 0; i < 2; ++i) {
            const bf16x* ap = &At[cur][(wm + i * 16 + m16) * 34];
#pragma unroll
            for (int v = 0; v < 8; ++v) afr[i][v] = *(const u32*)&ap[a_koff(v, lh)];
        }
#pragma unroll
        for (int j = 0; j < 4; ++j) {
            const bf16x* bp = &Bt[cur][wn + j * 16 + m16];
#pragma unroll
            for (int v = 0; v < 8; ++v) { const int kk = a_koff(v, lh); bfr[j][v] = (u32)bp[kk * 130] | ((u32)bp[(kk + 1) * 130] << 16); }
        }
#pragma unroll
        for (int i = 0; i < 2; ++i)
#pragma unroll
            for (int j = 0; j < 4; ++j)
                acc[i][j] = __builtin_amdgcn_wmma_f32_16x16x32_bf16(
                    false, __builtin_bit_cast(v16bf, afr[i]),
                    false, __builtin_bit_cast(v16bf, bfr[j]),
                    (short)0, acc[i][j], false, false);

        if (more) sstore(cur ^ 1);
        tdm_wait();
        __syncthreads();
        cur ^= 1;
    }

    // epilogue: lane&15 = N, vgpr r + 8*(lane>>4) = M
#pragma unroll
    for (int i = 0; i < 2; ++i)
#pragma unroll
        for (int j = 0; j < 4; ++j) {
            const int gn = bn + wn + j * 16 + m16;
            const float bv = bias ? bias[gn] : 0.0f;
#pragma unroll
            for (int r = 0; r < 8; ++r) {
                const int gm = bm + wm + i * 16 + r + 8 * lh;
                float vv = acc[i][j][r] + bv;
                if (flags & GEMM_FLAG_SILU) vv = vv / (1.0f + __expf(-vv));
                const size_t idx = (size_t)gm * N + gn;
                if (resid) vv += resid[idx];
                if (outF) outF[idx] = vv;
                if (outB) outB[idx] = f2bf(vv);
            }
        }
}

// ---------------- WMMA causal flash attention (V tile via TDM) ----------------
// grid: (S/64, H, B); block 128 = 4 waves; 64 queries/block (16/wave); 64-key tiles.
// Kt = K^T [hd][key] (B operand for Q*K^T), Vt = [key][hd] (already B layout for P*V).
__global__ __launch_bounds__(128) void k_attn(const bf16x* __restrict__ Q,
                                              const bf16x* __restrict__ K,
                                              const bf16x* __restrict__ V,
                                              bf16x* __restrict__ O) {
    __shared__ bf16x Kt[64][66];
    __shared__ bf16x Vt[64][66];    // stride 66 == TDM pad (128B rows + 4B pad)
    __shared__ bf16x Pt[4][16][66];

    const int qblk = blockIdx.x, h = blockIdx.y, b = blockIdx.z;
    const int w = threadIdx.x >> 5, lane = threadIdx.x & 31;
    const int m16 = lane & 15;
    const int lh  = lane >> 4;
    const size_t base = (size_t)b * Sn * Dm + (size_t)h * HDn;
    const int q0 = qblk * 64;
    const int qw = q0 + w * 16;

    // Q A-fragments kept in registers for the whole key loop (unscaled)
    v8u qa[2];
#pragma unroll
    for (int s2 = 0; s2 < 2; ++s2)
#pragma unroll
        for (int v = 0; v < 8; ++v)
            qa[s2][v] = *(const u32*)(Q + base + (size_t)(qw + m16) * Dm +
                                      a_koff(v, lh) + 32 * s2);

    v8f oacc[4];
#pragma unroll
    for (int f = 0; f < 4; ++f)
#pragma unroll
        for (int r = 0; r < 8; ++r) oacc[f][r] = 0.0f;
    float mrun[8], lrun[8];
#pragma unroll
    for (int r = 0; r < 8; ++r) { mrun[r] = -INFINITY; lrun[r] = 0.0f; }

    const int ntiles = qblk + 1;
    for (int kt = 0; kt < ntiles; ++kt) {
        const int kb = kt * 64;
        __syncthreads();
        if (threadIdx.x < 32)   // wave 0: V tile via TDM (straight 64x64 copy, padded)
            tdm_load_2d((u32)(size_t)&Vt[0][0], V + base + (size_t)kb * Dm,
                        64, 64, 64, 64, (u32)Dm, /*pad 128B+4B*/ 4, 0);
        // K tile: transpose on store (TDM cannot transpose)
        for (int c = threadIdx.x; c < 512; c += 128) {
            const int r = c >> 3, col = (c & 7) * 8;
            uint4 dk = *(const uint4*)(K + base + (size_t)(kb + r) * Dm + col);
            const bf16x* kp = (const bf16x*)&dk;
#pragma unroll
            for (int i = 0; i < 8; ++i) Kt[col + i][r] = kp[i];
#if !USE_TDM
            uint4 dv = *(const uint4*)(V + base + (size_t)(kb + r) * Dm + col);
            u32* vd = (u32*)&Vt[r][col];
            vd[0] = dv.x; vd[1] = dv.y; vd[2] = dv.z; vd[3] = dv.w;
#endif
        }
        tdm_wait();
        __syncthreads();

        // ---- S = Q K^T : 8 WMMAs ----
        v8f sc[4];
#pragma unroll
        for (int f = 0; f < 4; ++f)
#pragma unroll
            for (int r = 0; r < 8; ++r) sc[f][r] = 0.0f;
#pragma unroll
        for (int s2 = 0; s2 < 2; ++s2) {
#pragma unroll
            for (int f = 0; f < 4; ++f) {
                v8u kbf;
#pragma unroll
                for (int v = 0; v < 8; ++v) { const int kk = 32 * s2 + a_koff(v, lh);
                    kbf[v] = (u32)Kt[kk][f * 16 + m16] | ((u32)Kt[kk + 1][f * 16 + m16] << 16); }
                sc[f] = __builtin_amdgcn_wmma_f32_16x16x32_bf16(
                    false, __builtin_bit_cast(v16bf, qa[s2]),
                    false, __builtin_bit_cast(v16bf, kbf),
                    (short)0, sc[f], false, false);
            }
        }

        // ---- scale + causal mask + row max (rows live in lane-halves) ----
        float rmax[8];
#pragma unroll
        for (int r = 0; r < 8; ++r) rmax[r] = -INFINITY;
#pragma unroll
        for (int f = 0; f < 4; ++f) {
            const int key = kb + f * 16 + m16;
#pragma unroll
            for (int r = 0; r < 8; ++r) {
                const int qrow = qw + r + 8 * lh;
                float vv = sc[f][r] * 0.125f;       // 1/sqrt(HD)
                vv = (key <= qrow) ? vv : -INFINITY;
                sc[f][r] = vv;
                rmax[r] = fmaxf(rmax[r], vv);
            }
        }
#pragma unroll
        for (int d = 1; d <= 8; d <<= 1)
#pragma unroll
            for (int r = 0; r < 8; ++r)
                rmax[r] = fmaxf(rmax[r], __shfl_xor(rmax[r], d, 32));

        // ---- online softmax update ----
        float corr[8];
#pragma unroll
        for (int r = 0; r < 8; ++r) {
            const float mnew = fmaxf(mrun[r], rmax[r]);
            corr[r] = __expf(mrun[r] - mnew);
            mrun[r] = mnew;
        }
        float rsum[8];
#pragma unroll
        for (int r = 0; r < 8; ++r) rsum[r] = 0.0f;
#pragma unroll
        for (int f = 0; f < 4; ++f)
#pragma unroll
            for (int r = 0; r < 8; ++r) {
                const float p = __expf(sc[f][r] - mrun[r]);
                sc[f][r] = p;
                rsum[r] += p;
            }
#pragma unroll
        for (int d = 1; d <= 8; d <<= 1)
#pragma unroll
            for (int r = 0; r < 8; ++r) rsum[r] += __shfl_xor(rsum[r], d, 32);
#pragma unroll
        for (int r = 0; r < 8; ++r) lrun[r] = lrun[r] * corr[r] + rsum[r];
#pragma unroll
        for (int f = 0; f < 4; ++f)
#pragma unroll
            for (int r = 0; r < 8; ++r) oacc[f][r] *= corr[r];

        // ---- P: C layout -> A layout via per-wave LDS tile (wave-local) ----
#pragma unroll
        for (int f = 0; f < 4; ++f)
#pragma unroll
            for (int r = 0; r < 8; ++r)
                Pt[w][r + 8 * lh][f * 16 + m16] = f2bf(sc[f][r]);

        // ---- O += P V : 8 WMMAs ----
#pragma unroll
        for (int s2 = 0; s2 < 2; ++s2) {
            v8u pa;
#pragma unroll
            for (int v = 0; v < 8; ++v)
                pa[v] = *(const u32*)&Pt[w][m16][a_koff(v, lh) + 32 * s2];
#pragma unroll
            for (int f = 0; f < 4; ++f) {
                v8u vbf;
#pragma unroll
                for (int v = 0; v < 8; ++v) { const int kk = 32 * s2 + a_koff(v, lh);
                    vbf[v] = (u32)Vt[kk][f * 16 + m16] | ((u32)Vt[kk + 1][f * 16 + m16] << 16); }
                oacc[f] = __builtin_amdgcn_wmma_f32_16x16x32_bf16(
                    false, __builtin_bit_cast(v16bf, pa),
                    false, __builtin_bit_cast(v16bf, vbf),
                    (short)0, oacc[f], false, false);
            }
        }
    }

    // ---- normalize + store ----
#pragma unroll
    for (int f = 0; f < 4; ++f)
#pragma unroll
        for (int r = 0; r < 8; ++r) {
            const int qrow = qw + r + 8 * lh;
            O[base + (size_t)qrow * Dm + f * 16 + m16] =
                f2bf(oacc[f][r] / lrun[r]);
        }
}

// ---------------- host orchestration ----------------
extern "C" void kernel_launch(void* const* d_in, const int* in_sizes, int n_in,
                              void* d_out, int out_size, void* d_ws, size_t ws_size,
                              hipStream_t stream) {
    (void)in_sizes; (void)n_in; (void)out_size; (void)ws_size;

    const int*   tokens = (const int*)d_in[0];
    const float* emb    = (const float*)d_in[1];
    auto P = [&](int layer, int j) -> const float* {
        return (const float*)d_in[2 + 16 * layer + j];
    };
    const float* nfk  = (const float*)d_in[2 + 16 * NLn + 0];
    const float* nfb  = (const float*)d_in[2 + 16 * NLn + 1];
    const float* lm_w = (const float*)d_in[2 + 16 * NLn + 2];
    const float* lm_b = (const float*)d_in[2 + 16 * NLn + 3];
    const float* pos  = (const float*)d_in[2 + 16 * NLn + 4];

    char* ws = (char*)d_ws;
    size_t off = 0;
    auto alloc = [&](size_t bytes) -> char* {
        off = (off + 255) & ~(size_t)255;
        char* p = ws + off;
        off += bytes;
        return p;
    };

    const size_t szDD = (size_t)Dm * Dm;
    const size_t szDF = (size_t)DFFn * Dm;

    bf16x* wq[NLn]; bf16x* wk[NLn]; bf16x* wv[NLn]; bf16x* wo[NLn];
    bf16x* w1[NLn]; bf16x* w2[NLn];
    for (int lyr = 0; lyr < NLn; ++lyr) {
        wq[lyr] = (bf16x*)alloc(szDD * 2);
        wk[lyr] = (bf16x*)alloc(szDD * 2);
        wv[lyr] = (bf16x*)alloc(szDD * 2);
        wo[lyr] = (bf16x*)alloc(szDD * 2);
        w1[lyr] = (bf16x*)alloc(szDF * 2);
        w2[lyr] = (bf16x*)alloc(szDF * 2);
    }
    bf16x* lmw = (bf16x*)alloc((size_t)Vn * Dm * 2);

    float* X  = (float*)alloc((size_t)Mtok * Dm * 4);
    bf16x* Hb = (bf16x*)alloc((size_t)Mtok * Dm * 2);
    bf16x* Qb = (bf16x*)alloc((size_t)Mtok * Dm * 2);
    bf16x* Kb = (bf16x*)alloc((size_t)Mtok * Dm * 2);
    bf16x* Vb = (bf16x*)alloc((size_t)Mtok * Dm * 2);
    bf16x* Ab = (bf16x*)alloc((size_t)Mtok * Dm * 2);
    bf16x* Fb = (bf16x*)alloc((size_t)Mtok * DFFn * 2);

    auto cvt = [&](const float* src, bf16x* dst, size_t n) {
        int blocks = (int)((n + 255 * 8) / (256 * 8));
        if (blocks > 2048) blocks = 2048;
        if (blocks < 1) blocks = 1;
        k_cvt<<<blocks, 256, 0, stream>>>(src, dst, (int)n);
    };
    for (int lyr = 0; lyr < NLn; ++lyr) {
        cvt(P(lyr, 2),  wq[lyr], szDD);
        cvt(P(lyr, 4),  wk[lyr], szDD);
        cvt(P(lyr, 6),  wv[lyr], szDD);
        cvt(P(lyr, 8),  wo[lyr], szDD);
        cvt(P(lyr, 12), w1[lyr], szDF);
        cvt(P(lyr, 14), w2[lyr], szDF);
    }
    cvt(lm_w, lmw, (size_t)Vn * Dm);

    k_embed<<<2048, 256, 0, stream>>>(tokens, emb, pos, X);

    const dim3 gD(Dm / 128, Mtok / 128);
    const dim3 gF(DFFn / 128, Mtok / 128);
    const dim3 gL(Vn / 128, Mtok / 128);
    const dim3 gA(Sn / 64, Hn, Bn);

    for (int lyr = 0; lyr < NLn; ++lyr) {
        k_ln<<<Mtok, 256, 0, stream>>>(X, P(lyr, 0), P(lyr, 1), Hb);
        k_gemm<<<gD, 256, 0, stream>>>(Hb, wq[lyr], P(lyr, 3), nullptr, nullptr, Qb,
                                       Mtok, Dm, Dm, 0);
        k_gemm<<<gD, 256, 0, stream>>>(Hb, wk[lyr], P(lyr, 5), nullptr, nullptr, Kb,
                                       Mtok, Dm, Dm, 0);
        k_gemm<<<gD, 256, 0, stream>>>(Hb, wv[lyr], P(lyr, 7), nullptr, nullptr, Vb,
                                       Mtok, Dm, Dm, 0);
        k_attn<<<gA, 128, 0, stream>>>(Qb, Kb, Vb, Ab);
        k_gemm<<<gD, 256, 0, stream>>>(Ab, wo[lyr], P(lyr, 9), X, X, nullptr,
                                       Mtok, Dm, Dm, 0);
        k_ln<<<Mtok, 256, 0, stream>>>(X, P(lyr, 10), P(lyr, 11), Hb);
        k_gemm<<<gF, 256, 0, stream>>>(Hb, w1[lyr], P(lyr, 13), nullptr, nullptr, Fb,
                                       Mtok, DFFn, Dm, GEMM_FLAG_SILU);
        k_gemm<<<gD, 256, 0, stream>>>(Fb, w2[lyr], P(lyr, 15), X, X, nullptr,
                                       Mtok, Dm, DFFn, 0);
    }

    k_ln<<<Mtok, 256, 0, stream>>>(X, nfk, nfb, Hb);
    k_gemm<<<gL, 256, 0, stream>>>(Hb, lmw, lm_b, nullptr, (float*)d_out, nullptr,
                                   Mtok, Vn, Dm, 0);
}
